// AttnBlock_24429773979707
// MI455X (gfx1250) — matrix operand order repaired and checked by reading, not compiled, not run on hardware
//
#include <hip/hip_runtime.h>
#include <stdint.h>

typedef __attribute__((ext_vector_type(8)))  float   v8f;
typedef __attribute__((ext_vector_type(16))) __bf16  v16bf;
typedef __attribute__((ext_vector_type(8)))  __bf16  v8bf;

#define BATCH 4
#define CH    512
#define HW    4096
#define NGRP  32
#define CPG   16          // channels per group
#define SCALE 0.04419417382415922f   // 1/sqrt(512)

#if defined(__AMDGCN__) && __has_builtin(__builtin_amdgcn_global_load_async_to_lds_b128) && \
    __has_builtin(__builtin_amdgcn_s_wait_asynccnt)
#define HAS_ASYNC_LDS 1
typedef int v4i_vec __attribute__((vector_size(16)));   // matches builtin param type from diagnostic
#else
#define HAS_ASYNC_LDS 0
#endif

// ---------------- WMMA helpers (gfx1250, wave32) ----------------

__device__ __forceinline__ v8f wmma_bf16(v16bf a, v16bf b, v8f c) {
    // D = A(16x32) * B(32x16) + C(16x16 f32)
    return __builtin_amdgcn_wmma_f32_16x16x32_bf16(
        /*neg_a=*/false, a, /*neg_b=*/false, b,
        /*c_mod=*/(short)0, c, /*reuse_a=*/false, /*reuse_b=*/false);
}

// A fragment: 16x32 (MxK). lane m=lane&15 is row; half=lane>>4.
// per lane: K = half*8 .. half*8+7  then  16+half*8 .. 16+half*8+7
__device__ __forceinline__ v16bf load_a_frag(const __bf16* base, int ld, int lane) {
    const int m = lane & 15, half = lane >> 4;
    const __bf16* p = base + (size_t)m * ld + half * 8;
    v8bf lo = *(const v8bf*)(p);
    v8bf hi = *(const v8bf*)(p + 16);
    return __builtin_shufflevector(lo, hi, 0,1,2,3,4,5,6,7,8,9,10,11,12,13,14,15);
}

// B fragment: 32x16 (KxN). lane n=lane&15 is column; K = half*8 .. half*8+7 then 16+half*8 .. 16+half*8+7
__device__ __forceinline__ v16bf load_b_frag(const __bf16* base, int ld, int lane) {
    const int n = lane & 15, half = lane >> 4;
    const __bf16* p = base + (size_t)n * ld + half * 8;
    v8bf lo = *(const v8bf*)(p);
    v8bf hi = *(const v8bf*)(p + 16);
    return __builtin_shufflevector(lo, hi, 0,1,2,3,4,5,6,7,8,9,10,11,12,13,14,15);
}

__device__ __forceinline__ v8f zero_v8f() {
    v8f z = {0.f,0.f,0.f,0.f,0.f,0.f,0.f,0.f};
    return z;
}

// ---------------- Kernel 1: GroupNorm statistics ----------------
// one block per (b, group): 16 channels x 4096 pixels contiguous
__global__ void gn_stats_kernel(const float* __restrict__ x, float* __restrict__ stats) {
    __shared__ float s1[256], s2[256];
    const int bg = blockIdx.x;               // b*32+g
    const float* xp = x + (size_t)bg * (CPG * HW);
    float sum = 0.f, sq = 0.f;
    for (int i = threadIdx.x; i < CPG * HW; i += 256) {
        float v = xp[i];
        sum += v; sq += v * v;
    }
    s1[threadIdx.x] = sum; s2[threadIdx.x] = sq;
    __syncthreads();
    for (int s = 128; s > 0; s >>= 1) {
        if (threadIdx.x < s) { s1[threadIdx.x] += s1[threadIdx.x + s];
                               s2[threadIdx.x] += s2[threadIdx.x + s]; }
        __syncthreads();
    }
    if (threadIdx.x == 0) {
        const float inv_n = 1.0f / (float)(CPG * HW);
        float mean = s1[0] * inv_n;
        float var  = s2[0] * inv_n - mean * mean;
        stats[bg * 2 + 0] = mean;
        stats[bg * 2 + 1] = rsqrtf(var + 1e-6f);
    }
}

// ---------------- Kernel 2: apply GN + transpose to token-major bf16 ----------------
// x[b][c][n] (f32) -> h[b][n][c] (bf16), 64x64 LDS tile transpose
__global__ void gn_apply_kernel(const float* __restrict__ x, const float* __restrict__ stats,
                                const float* __restrict__ gamma, const float* __restrict__ beta,
                                __bf16* __restrict__ h) {
    __shared__ __bf16 tile[64][66];
    const int n0 = blockIdx.x * 64;
    const int c0 = blockIdx.y * 64;
    const int b  = blockIdx.z;
    const int nn = threadIdx.x & 63;
    const int cb = threadIdx.x >> 6;       // 0..3
    for (int i = 0; i < 16; ++i) {
        const int cc = cb + i * 4;
        const int c  = c0 + cc;
        const int g  = c >> 4;
        const float mean = stats[(b * NGRP + g) * 2 + 0];
        const float rstd = stats[(b * NGRP + g) * 2 + 1];
        const float xv = x[((size_t)(b * CH + c)) * HW + n0 + nn];
        tile[cc][nn] = (__bf16)((xv - mean) * rstd * gamma[c] + beta[c]);
    }
    __syncthreads();
    const int cc2 = threadIdx.x & 63;
    const int nb  = threadIdx.x >> 6;
    for (int i = 0; i < 16; ++i) {
        const int nn2 = nb + i * 4;
        h[((size_t)(b * HW + n0 + nn2)) * CH + c0 + cc2] = tile[cc2][nn2];
    }
}

// ---------------- Kernel 3: pack weights fp32 -> bf16 ----------------
__global__ void pack_bf16_kernel(const float* __restrict__ src, __bf16* __restrict__ dst, int n) {
    for (int i = blockIdx.x * blockDim.x + threadIdx.x; i < n; i += gridDim.x * blockDim.x)
        dst[i] = (__bf16)src[i];
}

// ---------------- Kernel 4: projection GEMM ----------------
// trans==0: out[b][tok][o] = sum_c h[b][tok][c] * w[o][c] + bias[o]   (Q, K)
// trans==1: out[b][o][tok] = sum_c w[o][c] * h[b][tok][c] + bias[o]   (V, channel-major)
// block = 8 waves; wave tile = 16 (M) x 64 (N); block tile = 128 x 64
__global__ void proj_gemm_kernel(const __bf16* __restrict__ hmat, const __bf16* __restrict__ wmat,
                                 const float* __restrict__ bias, __bf16* __restrict__ out,
                                 int trans) {
    const int b    = blockIdx.z;
    const int wv   = threadIdx.x >> 5;
    const int lane = threadIdx.x & 31;
    const int M0   = blockIdx.x * 128 + wv * 16;
    const int N0   = blockIdx.y * 64;

    const __bf16* Abase = trans ? (wmat + (size_t)M0 * CH)
                                : (hmat + ((size_t)(b * HW + M0)) * CH);
    v8f acc[4];
    for (int nt = 0; nt < 4; ++nt) acc[nt] = zero_v8f();

    for (int kk = 0; kk < CH / 32; ++kk) {
        v16bf a = load_a_frag(Abase + kk * 32, CH, lane);
        #pragma unroll
        for (int nt = 0; nt < 4; ++nt) {
            const __bf16* Bbase = trans
                ? (hmat + ((size_t)(b * HW + N0 + nt * 16)) * CH + kk * 32)
                : (wmat + ((size_t)(N0 + nt * 16)) * CH + kk * 32);
            v16bf bf = load_b_frag(Bbase, CH, lane);
            acc[nt] = wmma_bf16(a, bf, acc[nt]);
        }
    }
    const int half = lane >> 4, n = lane & 15;
    #pragma unroll
    for (int nt = 0; nt < 4; ++nt) {
        #pragma unroll
        for (int r = 0; r < 8; ++r) {
            const int mrow = half * 8 + r;
            float v = acc[nt][r];
            if (trans) {
                const int o = M0 + mrow;
                out[((size_t)(b * CH + o)) * HW + N0 + nt * 16 + n] = (__bf16)(v + bias[o]);
            } else {
                const int o = N0 + nt * 16 + n;
                out[((size_t)(b * HW + M0 + mrow)) * CH + o] = (__bf16)(v + bias[o]);
            }
        }
    }
}

// ---------------- Kernel 5: attention (S row-block resident in 320KB LDS) ----------------
// block = 16 query rows, 8 waves. LDS: S[16][4096] f32 (256KB) + Q[16][512] bf16 (16KB) + reductions
#define ATTN_SMEM (16 * HW * 4 + 16 * CH * 2 + (256 + 16 + 16) * 4)
__global__ void attn_kernel(const __bf16* __restrict__ q, const __bf16* __restrict__ k,
                            const __bf16* __restrict__ vT, __bf16* __restrict__ O) {
    extern __shared__ char smem[];
    float*  S      = (float*)smem;                               // [16][4096]
    __bf16* Qs     = (__bf16*)(smem + 16 * HW * 4);              // [16][512]
    float*  red    = (float*)(smem + 16 * HW * 4 + 16 * CH * 2); // [16][16]
    float*  rowmax = red + 256;                                  // [16]
    float*  invs   = rowmax + 16;                                // [16]

    const int b  = blockIdx.y;
    const int q0 = blockIdx.x * 16;
    const int t  = threadIdx.x;
    const int wv = t >> 5;
    const int lane = t & 31;
    const int m = lane & 15, half = lane >> 4, n = lane & 15;

    // stage Q tile into LDS (16 rows x 512 bf16, contiguous = 1024 x 16B)
#if HAS_ASYNC_LDS
    {
        const __bf16* src = q + ((size_t)(b * HW + q0)) * CH;
        #pragma unroll
        for (int j = 0; j < 4; ++j) {
            const int i = t + 256 * j;          // 1024 b128 transfers
            __builtin_amdgcn_global_load_async_to_lds_b128(
                (v4i_vec*)(src + i * 8),
                (v4i_vec*)(Qs + i * 8),
                0, 0);
        }
        __builtin_amdgcn_s_wait_asynccnt(0);
    }
#else
    {
        const uint32_t* src = (const uint32_t*)(q + ((size_t)(b * HW + q0)) * CH);
        uint32_t* dst = (uint32_t*)Qs;
        for (int i = t; i < 16 * CH / 2; i += 256) dst[i] = src[i];
    }
#endif
    __syncthreads();

    // ---- phase 1: S = Q K^T. wave wv owns keys [wv*512, wv*512+512) ----
    v16bf afr[16];
    #pragma unroll
    for (int kk = 0; kk < 16; ++kk)
        afr[kk] = load_a_frag(Qs + kk * 32, CH, lane);   // Q stays in registers

    #pragma unroll 1
    for (int kt = 0; kt < 32; ++kt) {
        const int key0 = wv * 512 + kt * 16;
        if (kt + 2 < 32) {   // deep prefetch of next-next K tile row (per-lane)
            const int keyp = wv * 512 + (kt + 2) * 16;
            __builtin_prefetch(k + ((size_t)(b * HW + keyp + n)) * CH + half * 256, 0, 1);
        }
        v8f acc = zero_v8f();
        #pragma unroll
        for (int kk = 0; kk < 16; ++kk) {
            v16bf bf = load_b_frag(k + ((size_t)(b * HW + key0)) * CH + kk * 32, CH, lane);
            acc = wmma_bf16(afr[kk], bf, acc);
        }
        #pragma unroll
        for (int r = 0; r < 8; ++r)
            S[(half * 8 + r) * HW + key0 + n] = acc[r];
    }
    __syncthreads();

    // ---- phase 2: softmax over each of 16 rows (16 threads per row) ----
    {
        const int row = t >> 4, sub = t & 15;
        float* Srow = S + row * HW;
        float mx = -3.402823466e+38f;
        #pragma unroll 1
        for (int j = sub; j < HW; j += 16) mx = fmaxf(mx, Srow[j] * SCALE);
        red[row * 16 + sub] = mx;
        __syncthreads();
        if (sub == 0) {
            float r2 = red[row * 16];
            #pragma unroll
            for (int i = 1; i < 16; ++i) r2 = fmaxf(r2, red[row * 16 + i]);
            rowmax[row] = r2;
        }
        __syncthreads();
        const float rm = rowmax[row];
        float sum = 0.f;
        #pragma unroll 1
        for (int j = sub; j < HW; j += 16) {
            float e = __expf(Srow[j] * SCALE - rm);
            Srow[j] = e;                       // store unnormalized exp
            sum += e;
        }
        red[row * 16 + sub] = sum;
        __syncthreads();
        if (sub == 0) {
            float s2 = 0.f;
            #pragma unroll
            for (int i = 0; i < 16; ++i) s2 += red[row * 16 + i];
            invs[row] = 1.0f / s2;
        }
    }
    __syncthreads();

    // ---- phase 3: O = P V. wave wv owns d-slice [wv*64, wv*64+64) ----
    const int d0 = wv * 64;
    v8f oacc[4];
    for (int nt = 0; nt < 4; ++nt) oacc[nt] = zero_v8f();
    const float* Sm = S + m * HW;
    const float invm = invs[m];

    #pragma unroll 1
    for (int kt = 0; kt < HW / 32; ++kt) {
        if (kt + 2 < HW / 32) {   // prefetch next-next V^T key segment (per-lane: d-rows d0..d0+31 and d0+32..d0+63)
            __builtin_prefetch(vT + ((size_t)(b * CH + d0 + lane)) * HW + (kt + 2) * 32, 0, 1);
            __builtin_prefetch(vT + ((size_t)(b * CH + d0 + 32 + lane)) * HW + (kt + 2) * 32, 0, 1);
        }
        const int base = kt * 32 + half * 8;
        float4 f0 = *(const float4*)(Sm + base);
        float4 f1 = *(const float4*)(Sm + base + 4);
        float4 f2 = *(const float4*)(Sm + base + 16);
        float4 f3 = *(const float4*)(Sm + base + 20);
        v16bf a;
        a[0]=(__bf16)(f0.x*invm); a[1]=(__bf16)(f0.y*invm); a[2]=(__bf16)(f0.z*invm); a[3]=(__bf16)(f0.w*invm);
        a[4]=(__bf16)(f1.x*invm); a[5]=(__bf16)(f1.y*invm); a[6]=(__bf16)(f1.z*invm); a[7]=(__bf16)(f1.w*invm);
        a[8]=(__bf16)(f2.x*invm); a[9]=(__bf16)(f2.y*invm); a[10]=(__bf16)(f2.z*invm); a[11]=(__bf16)(f2.w*invm);
        a[12]=(__bf16)(f3.x*invm); a[13]=(__bf16)(f3.y*invm); a[14]=(__bf16)(f3.z*invm); a[15]=(__bf16)(f3.w*invm);
        #pragma unroll
        for (int nt = 0; nt < 4; ++nt) {
            v16bf bf = load_b_frag(vT + ((size_t)(b * CH + d0 + nt * 16)) * HW + kt * 32, HW, lane);
            oacc[nt] = wmma_bf16(a, bf, oacc[nt]);
        }
    }
    #pragma unroll
    for (int nt = 0; nt < 4; ++nt)
        #pragma unroll
        for (int r = 0; r < 8; ++r)
            O[((size_t)(b * HW + q0 + half * 8 + r)) * CH + d0 + nt * 16 + n] = (__bf16)oacc[nt][r];
}

// ---------------- Kernel 6: output projection + bias + residual (fp32 out) ----------------
// y[b][o][hw] = x[b][o][hw] + bo[o] + sum_c wo[o][c] * Oattn[b][hw][c]
__global__ void out_proj_kernel(const __bf16* __restrict__ Oa, const __bf16* __restrict__ wob,
                                const float* __restrict__ bo, const float* __restrict__ x,
                                float* __restrict__ y) {
    const int b    = blockIdx.z;
    const int wv   = threadIdx.x >> 5;
    const int lane = threadIdx.x & 31;
    const int M0   = blockIdx.x * 128 + wv * 16;   // o rows
    const int N0   = blockIdx.y * 64;              // hw cols

    const __bf16* Abase = wob + (size_t)M0 * CH;
    v8f acc[4];
    for (int nt = 0; nt < 4; ++nt) acc[nt] = zero_v8f();

    for (int kk = 0; kk < CH / 32; ++kk) {
        v16bf a = load_a_frag(Abase + kk * 32, CH, lane);
        #pragma unroll
        for (int nt = 0; nt < 4; ++nt) {
            v16bf bf = load_b_frag(Oa + ((size_t)(b * HW + N0 + nt * 16)) * CH + kk * 32, CH, lane);
            acc[nt] = wmma_bf16(a, bf, acc[nt]);
        }
    }
    const int half = lane >> 4, n = lane & 15;
    #pragma unroll
    for (int nt = 0; nt < 4; ++nt) {
        #pragma unroll
        for (int r = 0; r < 8; ++r) {
            const int o  = M0 + half * 8 + r;
            const int hw = N0 + nt * 16 + n;
            const size_t idx = ((size_t)(b * CH + o)) * HW + hw;
            y[idx] = x[idx] + bo[o] + acc[nt][r];
        }
    }
}

// ---------------- host launch ----------------
extern "C" void kernel_launch(void* const* d_in, const int* in_sizes, int n_in,
                              void* d_out, int out_size, void* d_ws, size_t ws_size,
                              hipStream_t stream) {
    const float* x     = (const float*)d_in[0];
    const float* gamma = (const float*)d_in[1];
    const float* beta  = (const float*)d_in[2];
    const float* wq    = (const float*)d_in[3];
    const float* bq    = (const float*)d_in[4];
    const float* wk    = (const float*)d_in[5];
    const float* bk    = (const float*)d_in[6];
    const float* wv    = (const float*)d_in[7];
    const float* bv    = (const float*)d_in[8];
    const float* wo    = (const float*)d_in[9];
    const float* bo    = (const float*)d_in[10];
    float* y = (float*)d_out;

    char* ws = (char*)d_ws;
    size_t off = 0;
    float*  stats = (float*)(ws + off);  off += 4096;
    const size_t TENS = (size_t)BATCH * HW * CH * sizeof(__bf16);   // 16.78 MB
    __bf16* h   = (__bf16*)(ws + off);   off += TENS;
    __bf16* qb  = (__bf16*)(ws + off);   off += TENS;
    __bf16* kb  = (__bf16*)(ws + off);   off += TENS;
    __bf16* vT  = (__bf16*)(ws + off);   off += TENS;   // [B][C][HW]
    __bf16* Oa  = (__bf16*)(ws + off);   off += TENS;
    const size_t WSZ = (size_t)CH * CH * sizeof(__bf16);
    __bf16* wqb = (__bf16*)(ws + off);   off += WSZ;
    __bf16* wkb = (__bf16*)(ws + off);   off += WSZ;
    __bf16* wvb = (__bf16*)(ws + off);   off += WSZ;
    __bf16* wob = (__bf16*)(ws + off);   off += WSZ;

    (void)hipFuncSetAttribute((const void*)attn_kernel,
                              hipFuncAttributeMaxDynamicSharedMemorySize, ATTN_SMEM);

    gn_stats_kernel<<<dim3(BATCH * NGRP), 256, 0, stream>>>(x, stats);
    gn_apply_kernel<<<dim3(HW / 64, CH / 64, BATCH), 256, 0, stream>>>(x, stats, gamma, beta, h);

    pack_bf16_kernel<<<512, 256, 0, stream>>>(wq, wqb, CH * CH);
    pack_bf16_kernel<<<512, 256, 0, stream>>>(wk, wkb, CH * CH);
    pack_bf16_kernel<<<512, 256, 0, stream>>>(wv, wvb, CH * CH);
    pack_bf16_kernel<<<512, 256, 0, stream>>>(wo, wob, CH * CH);

    // Q, K: token-major [B][HW][C]
    proj_gemm_kernel<<<dim3(HW / 128, CH / 64, BATCH), 256, 0, stream>>>(h, wqb, bq, qb, 0);
    proj_gemm_kernel<<<dim3(HW / 128, CH / 64, BATCH), 256, 0, stream>>>(h, wkb, bk, kb, 0);
    // V: channel-major [B][C][HW] so PV B-fragments are contiguous
    proj_gemm_kernel<<<dim3(CH / 128, HW / 64, BATCH), 256, 0, stream>>>(h, wvb, bv, vT, 1);

    attn_kernel<<<dim3(HW / 16, BATCH), 256, ATTN_SMEM, stream>>>(qb, kb, vT, Oa);

    out_proj_kernel<<<dim3(CH / 128, HW / 64, BATCH), 256, 0, stream>>>(Oa, wob, bo, x, y);
}
